// AllegroEmbedding_12876311954156
// MI455X (gfx1250) — matrix operand order repaired and checked by reading, not compiled, not run on hardware
//
#include <hip/hip_runtime.h>
#include <hip/hip_bf16.h>
#include <math.h>

// ---------------------------------------------------------------------------
// Types for CDNA5 WMMA
// ---------------------------------------------------------------------------
typedef __attribute__((ext_vector_type(16))) __bf16 v16bf;
typedef __attribute__((ext_vector_type(8)))  __bf16 v8bf;
typedef __attribute__((ext_vector_type(8)))  float  v8f;

enum { F_SILU = 1, F_MULSW = 2, F_ACC = 4 };

#define SQ3f  1.7320508075688772f
#define SQ5f  2.23606797749979f
#define SQ15f 3.872983346207417f

// ---------------------------------------------------------------------------
// Spherical-harmonic polynomial tables (l<=2, real SH as used by reference)
// ---------------------------------------------------------------------------
__constant__ int SH_NM[9] = {1,1,1,1,1,1,2,1,2};
__constant__ signed char SH_EX[9][2][3] = {
  {{0,0,0},{0,0,0}},
  {{0,1,0},{0,0,0}},
  {{0,0,1},{0,0,0}},
  {{1,0,0},{0,0,0}},
  {{1,1,0},{0,0,0}},
  {{0,1,1},{0,0,0}},
  {{0,0,2},{0,0,0}},
  {{1,0,1},{0,0,0}},
  {{2,0,0},{0,2,0}},
};
__constant__ float SH_CO[9][2] = {
  {1.f, 0.f}, {SQ3f, 0.f}, {SQ3f, 0.f}, {SQ3f, 0.f},
  {SQ15f, 0.f}, {SQ15f, 0.f}, {1.5f*SQ5f, -0.5f*SQ5f},
  {SQ15f, 0.f}, {0.5f*SQ15f, -0.5f*SQ15f},
};
// (l1,l2,l3) -> path index in lexicographic nonzero-block order (NPATHS=11)
__constant__ int PATH_LUT[27] = {
   0,-1,-1, -1, 1,-1, -1,-1, 2,
  -1, 3,-1,  4,-1, 5, -1, 6,-1,
  -1,-1, 7, -1, 8,-1,  9,-1,10 };

__device__ __forceinline__ int ldeg(int i) { return (i == 0) ? 0 : (i < 4 ? 1 : 2); }

__device__ __forceinline__ float dfacf(int n) {
  float r = 1.f;
  while (n > 1) { r *= (float)n; n -= 2; }
  return r;
}
__device__ __forceinline__ float mono_int(int a, int b, int c) {
  if ((a | b | c) & 1) return 0.f;
  return dfacf(a - 1) * dfacf(b - 1) * dfacf(c - 1) / dfacf(a + b + c + 1);
}

// ---------------------------------------------------------------------------
// Build sparse CG table + per-layer per-channel path weights, on device.
// meta[0]=nnz, meta[1..10]=koff[0..9] (sorted by output index k), meta[11..]=packed ijk
// w[l*729*16 + t*16 + c] = tp_w[l][path(ijk)][c] * CG[ijk]
// ---------------------------------------------------------------------------
__global__ void cg_setup_kernel(const float* __restrict__ tp0,
                                const float* __restrict__ tp1,
                                const float* __restrict__ tp2,
                                int* __restrict__ meta, float* __restrict__ w) {
  __shared__ float cg[729];
  __shared__ int nnz_s;
  for (int t = threadIdx.x; t < 729; t += blockDim.x) {
    int i = t / 81, j = (t / 9) % 9, k = t % 9;
    float s = 0.f;
    for (int a = 0; a < SH_NM[i]; ++a)
      for (int b = 0; b < SH_NM[j]; ++b)
        for (int c = 0; c < SH_NM[k]; ++c) {
          int ax = SH_EX[i][a][0] + SH_EX[j][b][0] + SH_EX[k][c][0];
          int ay = SH_EX[i][a][1] + SH_EX[j][b][1] + SH_EX[k][c][1];
          int az = SH_EX[i][a][2] + SH_EX[j][b][2] + SH_EX[k][c][2];
          s += SH_CO[i][a] * SH_CO[j][b] * SH_CO[k][c] * mono_int(ax, ay, az);
        }
    cg[t] = s;
  }
  __syncthreads();
  if (threadIdx.x == 0) {
    int n = 0;
    meta[1] = 0;
    for (int k = 0; k < 9; ++k) {
      for (int t = 0; t < 729; ++t)
        if ((t % 9) == k && fabsf(cg[t]) > 1e-9f) meta[11 + n++] = t;
      meta[2 + k] = n;
    }
    meta[0] = n;
    nnz_s = n;
  }
  __syncthreads();
  int nnz = nnz_s;
  const float* tps[3] = {tp0, tp1, tp2};
  for (int u = threadIdx.x; u < nnz * 48; u += blockDim.x) {
    int l = u / (nnz * 16);
    int t = (u / 16) % nnz;
    int c = u % 16;
    int p = meta[11 + t];
    int i = p / 81, j = (p / 9) % 9, k = p % 9;
    int pid = PATH_LUT[ldeg(i) * 9 + ldeg(j) * 3 + ldeg(k)];
    w[(size_t)l * (729 * 16) + t * 16 + c] = tps[l][pid * 16 + c] * cg[p];
  }
}

// ---------------------------------------------------------------------------
// Species embedding gather: enc[n][c] = table[species[n]][c]
// ---------------------------------------------------------------------------
__global__ void enc_kernel(const float* __restrict__ table, const int* __restrict__ species,
                           float* __restrict__ enc, int N) {
  int t = blockIdx.x * blockDim.x + threadIdx.x;
  if (t >= N * 16) return;
  int n = t >> 4, c = t & 15;
  enc[t] = table[species[n] * 16 + c];
}

// ---------------------------------------------------------------------------
// Per-edge geometry: radial basis + spherical harmonics + feature concat
// feat layout: [enc_src(16) | enc_dst(16) | rb(8)] with ld=48
// ---------------------------------------------------------------------------
__global__ void edge_geom_kernel(const float* __restrict__ vec, const float* __restrict__ dist,
                                 const float* __restrict__ enc,
                                 const int* __restrict__ esrc, const int* __restrict__ edst,
                                 float* __restrict__ feat, float* __restrict__ Y, int E) {
  int e = blockIdx.x * blockDim.x + threadIdx.x;
  if (e >= E) return;
  float d = dist[e];
  float inv = 1.f / d;
  float x = vec[e * 3 + 0] * inv, y = vec[e * 3 + 1] * inv, z = vec[e * 3 + 2] * inv;
  float* Ye = Y + (size_t)e * 9;
  Ye[0] = 1.f;
  Ye[1] = SQ3f * y;  Ye[2] = SQ3f * z;  Ye[3] = SQ3f * x;
  Ye[4] = SQ15f * x * y;  Ye[5] = SQ15f * y * z;
  Ye[6] = 0.5f * SQ5f * (3.f * z * z - 1.f);
  Ye[7] = SQ15f * x * z;
  Ye[8] = 0.5f * SQ15f * (x * x - y * y);
  float* fe = feat + (size_t)e * 48;
  const float* es = enc + (size_t)esrc[e] * 16;
  const float* ed = enc + (size_t)edst[e] * 16;
#pragma unroll
  for (int c = 0; c < 16; ++c) { fe[c] = es[c]; fe[16 + c] = ed[c]; }
  float amp = sqrtf(2.f / 5.f);
  float ph = 3.14159265358979323846f * d * (1.f / 5.f);
#pragma unroll
  for (int n = 1; n <= 8; ++n) fe[31 + n] = amp * sinf((float)n * ph) * inv;
}

// ---------------------------------------------------------------------------
// bf16 WMMA GEMM: Out[M x N] = epilogue(A[M x Ktot] @ W[Ktot x N] + bias)
//
// A staged as fp32 in LDS; fragments convert fp32->bf16 in registers.
// Fast path (no concat, Ktot==Kpad<=lda1): each wave stages its OWN 16 rows
//   with per-lane GLOBAL_LOAD_ASYNC_TO_LDS_B128 (ASYNCcnt), so the K loop has
//   no block-wide barriers. Rows past M are clamped (outputs masked at store).
// Slow path (K=40 two-body / K=144 concat): cooperative barriered staging.
// W staged transposed [n][k] in bf16 LDS once per block.
// 8 waves/block, wave = 16 rows x (NT*16) cols, K step 32.
// ---------------------------------------------------------------------------
template <int NT, int FLAGS>
__global__ __launch_bounds__(256, 2)
void gemm_bf16_wmma(const float* __restrict__ A1, int lda1,
                    const float* __restrict__ A2, int lda2, int splitK,
                    int Ktot, int Kpad,
                    const float* __restrict__ W,
                    const float* __restrict__ bias,
                    const float* __restrict__ sw,
                    float* __restrict__ Out, int ldOut, int M) {
  constexpr int N = NT * 16;
  __shared__ __align__(16) float  Af[128][36];    // fp32 A tile, row = 144B (16B aligned)
  __shared__ __align__(16) __bf16 Wlds[N][168];   // transposed bf16 weights, KMAX=160 + pad

  const int tid  = threadIdx.x;
  const int lane = tid & 31;
  const int wv   = tid >> 5;
  const int half = lane >> 4;     // 0: lanes 0-15, 1: lanes 16-31
  const int r16  = lane & 15;
  const int eBase = blockIdx.x * 128;

  const bool fast = (A2 == nullptr) && (Ktot == Kpad) && (lda1 >= Kpad);

  // stage W (transposed, zero-padded to Kpad) once per block
  for (int u = tid; u < Kpad * N; u += 256) {
    int k = u / N, n = u % N;
    float v = (k < Ktot) ? W[k * N + n] : 0.f;
    Wlds[n][k] = (__bf16)v;
  }

  // per-thread async staging addresses: thread stages 16 floats (64B) of row tid>>1,
  // segment tid&1. Wave w thus stages exactly rows 16w..16w+15 == its A fragment rows.
  const int srow = tid >> 1;
  const int sseg = tid & 1;
  int erow = eBase + srow;
  if (erow >= M) erow = M - 1;                      // clamp: masked at store anyway
  const unsigned lds_base =
      (unsigned)(unsigned long long)(const void*)&Af[srow][sseg * 16];
  const unsigned long long gbase =
      (unsigned long long)(const void*)(A1 + (size_t)erow * lda1 + sseg * 16);

  v8f acc[NT];
#pragma unroll
  for (int t = 0; t < NT; ++t)
#pragma unroll
    for (int u = 0; u < 8; ++u) acc[t][u] = 0.f;

  __syncthreads();   // Wlds ready; Af about to be populated

  for (int kk = 0; kk < Kpad; kk += 32) {
    if (fast) {
      // async DMA global->LDS for this wave's own 16 rows; wait own ASYNCcnt.
      unsigned long long ga = gbase + (unsigned long long)kk * 4ull;
      asm volatile(
          "s_wait_dscnt 0x0\n\t"   // prior frag ds_loads done before overwrite
          "global_load_async_to_lds_b128 %0, %1, off\n\t"
          "global_load_async_to_lds_b128 %0, %1, off offset:16\n\t"
          "global_load_async_to_lds_b128 %0, %1, off offset:32\n\t"
          "global_load_async_to_lds_b128 %0, %1, off offset:48\n\t"
          "s_wait_asynccnt 0x0"
          :: "v"(lds_base), "v"(ga)
          : "memory");
    } else {
      __syncthreads();   // protect previous iteration's reads
      for (int u = tid; u < 128 * 32; u += 256) {
        int rr = u >> 5;
        int k  = kk + (u & 31);
        int e  = eBase + rr;
        float v = 0.f;
        if (e < M) {
          if (k < splitK)      v = A1[(size_t)e * lda1 + k];
          else if (k < Ktot)   v = A2[(size_t)e * lda2 + (k - splitK)];
        }
        Af[rr][u & 31] = v;
      }
      __syncthreads();
    }

    // A fragment (16x32): lanes 0-15 row M=lane K{0..7,16..23}; lanes 16-31 K{8..15,24..31}
    const float* arow = &Af[wv * 16 + r16][0];
    float4 q0 = *(const float4*)(arow + half * 8);
    float4 q1 = *(const float4*)(arow + half * 8 + 4);
    float4 q2 = *(const float4*)(arow + 16 + half * 8);
    float4 q3 = *(const float4*)(arow + 16 + half * 8 + 4);
    v16bf a;
    a[0]  = (__bf16)q0.x; a[1]  = (__bf16)q0.y; a[2]  = (__bf16)q0.z; a[3]  = (__bf16)q0.w;
    a[4]  = (__bf16)q1.x; a[5]  = (__bf16)q1.y; a[6]  = (__bf16)q1.z; a[7]  = (__bf16)q1.w;
    a[8]  = (__bf16)q2.x; a[9]  = (__bf16)q2.y; a[10] = (__bf16)q2.z; a[11] = (__bf16)q2.w;
    a[12] = (__bf16)q3.x; a[13] = (__bf16)q3.y; a[14] = (__bf16)q3.z; a[15] = (__bf16)q3.w;

#pragma unroll
    for (int t = 0; t < NT; ++t) {
      // B fragment (32x16 bf16): N=lane&15; lanes 0-15 K{0..7,16..23}, lanes 16-31 K{8..15,24..31}
      const __bf16* brow = &Wlds[t * 16 + r16][kk + half * 8];
      v8bf blo = *(const v8bf*)(brow);
      v8bf bhi = *(const v8bf*)(brow + 16);
      v16bf b;
#pragma unroll
      for (int u = 0; u < 8; ++u) { b[u] = blo[u]; b[u + 8] = bhi[u]; }
      acc[t] = __builtin_amdgcn_wmma_f32_16x16x32_bf16(
          false, a, false, b, (short)0, acc[t], false, false);
    }
  }

  // epilogue: C layout VGPR v -> M = v + half*8, N = t*16 + (lane&15)
#pragma unroll
  for (int t = 0; t < NT; ++t) {
#pragma unroll
    for (int v = 0; v < 8; ++v) {
      int m = v + half * 8;
      int e = eBase + wv * 16 + m;
      if (e < M) {
        int n = t * 16 + r16;
        float val = acc[t][v];
        if (bias) val += bias[n];
        if (FLAGS & F_SILU) val = val / (1.f + __expf(-val));
        if (FLAGS & F_MULSW) val *= sw[e];
        float* o = Out + (size_t)e * ldOut + n;
        if (FLAGS & F_ACC) val += *o;
        *o = val;
      }
    }
  }
}

// ---------------------------------------------------------------------------
// Vij init: Vij[e,c,i] = Y[e,i] * mix0[LDEG[i],c] * xw[e,c]
// ---------------------------------------------------------------------------
__global__ void vij_init_kernel(const float* __restrict__ Y, const float* __restrict__ mix0,
                                const float* __restrict__ xw, float* __restrict__ Vij, int E) {
  int t = blockIdx.x * blockDim.x + threadIdx.x;
  if (t >= E * 16) return;
  int e = t >> 4, c = t & 15;
  float xv = xw[(size_t)e * 16 + c];
  const float* Ye = Y + (size_t)e * 9;
  float* vp = Vij + (size_t)e * 144 + c * 9;
#pragma unroll
  for (int i = 0; i < 9; ++i) {
    int l = (i == 0) ? 0 : (i < 4 ? 1 : 2);
    vp[i] = Ye[i] * mix0[l * 16 + c] * xv;
  }
}

__global__ void zero_kernel(float* __restrict__ p, long n) {
  long i = (long)blockIdx.x * blockDim.x + threadIdx.x;
  if (i < n) p[i] = 0.f;
}

// rho scatter: density[src[e], c, i] += rho_pre[e,c] * Y[e,i]
__global__ void scatter_kernel(const float* __restrict__ rho_pre, const float* __restrict__ Y,
                               const int* __restrict__ esrc, float* __restrict__ density, int E) {
  int t = blockIdx.x * blockDim.x + threadIdx.x;
  if (t >= E * 16) return;
  int e = t >> 4, c = t & 15;
  float r = rho_pre[(size_t)e * 16 + c];
  const float* Ye = Y + (size_t)e * 9;
  float* drow = density + (size_t)esrc[e] * 144 + c * 9;
#pragma unroll
  for (int i = 0; i < 9; ++i) atomicAdd(drow + i, r * Ye[i]);
}

// ---------------------------------------------------------------------------
// Sparse CG contraction: Lij[e,c,k] = sum_{nz (i,j,k)} w[c] * Vij[e,c,i] * D[src,c,j]
// 16 edges x 16 channels per block. V/D staged per-thread in LDS (dynamic index).
// ---------------------------------------------------------------------------
#define MAXNNZ 512
__global__ __launch_bounds__(256, 2)
void lij_kernel(const float* __restrict__ Vij, const float* __restrict__ density,
                const int* __restrict__ esrc, const int* __restrict__ meta,
                const float* __restrict__ w, float* __restrict__ Lij,
                float* __restrict__ scals, int E) {
  __shared__ float wl[MAXNNZ * 16];
  __shared__ int   il[MAXNNZ];
  __shared__ int   koff[10];
  __shared__ float VD[256][19];
  const int tid = threadIdx.x;
  if (tid < 10) koff[tid] = meta[1 + tid];
  __syncthreads();
  int nnz = koff[9];
  if (nnz > MAXNNZ) nnz = MAXNNZ;
  for (int u = tid; u < nnz; u += 256) il[u] = meta[11 + u];
  for (int u = tid; u < nnz * 16; u += 256) wl[u] = w[u];
  __syncthreads();

  int e = blockIdx.x * 16 + (tid >> 4);
  int c = tid & 15;
  if (e >= E) return;

  float* vd = VD[tid];
  const float* vp = Vij + (size_t)e * 144 + c * 9;
  const float* dp = density + (size_t)esrc[e] * 144 + c * 9;
#pragma unroll
  for (int i = 0; i < 9; ++i) { vd[i] = vp[i]; vd[9 + i] = dp[i]; }

  float* lp = Lij + (size_t)e * 144 + c * 9;
#pragma unroll
  for (int k = 0; k < 9; ++k) {
    float s = 0.f;
    int t0 = koff[k], t1 = koff[k + 1];
    for (int t = t0; t < t1; ++t) {
      int p = il[t];
      s += wl[t * 16 + c] * vd[p / 81] * vd[9 + (p / 9) % 9];
    }
    lp[k] = s;
    if (k == 0) scals[(size_t)e * 16 + c] = s;
  }
}

// Vij_new[e,o,i] = sum_c Lij[e,c,i] * cm_W[o,c]
__global__ void vij_update_kernel(const float* __restrict__ Lij, const float* __restrict__ cmW,
                                  float* __restrict__ Vij, int E) {
  __shared__ float cm[256];
  for (int u = threadIdx.x; u < 256; u += blockDim.x) cm[u] = cmW[u];
  __syncthreads();
  int t = blockIdx.x * blockDim.x + threadIdx.x;
  if (t >= E * 9) return;
  int e = t / 9, i = t % 9;
  float L[16];
  const float* lp = Lij + (size_t)e * 144 + i;
#pragma unroll
  for (int c = 0; c < 16; ++c) L[c] = lp[c * 9];
  float* vp = Vij + (size_t)e * 144 + i;
#pragma unroll
  for (int o = 0; o < 16; ++o) {
    float s = 0.f;
#pragma unroll
    for (int c = 0; c < 16; ++c) s += L[c] * cm[o * 16 + c];
    vp[o * 9] = s;
  }
}

// ---------------------------------------------------------------------------
// Host launch
// ---------------------------------------------------------------------------
extern "C" void kernel_launch(void* const* d_in, const int* in_sizes, int n_in,
                              void* d_out, int out_size, void* d_ws, size_t ws_size,
                              hipStream_t stream) {
  (void)n_in; (void)out_size; (void)ws_size;
  const int *species, *esrc, *edst;
  const float *vecp, *dist, *sw;
  const float *stab, *tbW0, *tbW1, *tbb0, *tbb1, *Wv, *mix0;
  const float *embW[3], *embb[3], *tpw[3], *latW1[3], *latb1[3], *latW2[3], *latb2[3], *cmW[3];
  int N, E;

  if (in_sizes[0] != in_sizes[1]) {
    // Order A: top-level + nested dicts flattened in insertion order
    species = (const int*)d_in[0];
    esrc    = (const int*)d_in[1];
    edst    = (const int*)d_in[2];
    vecp    = (const float*)d_in[3];
    dist    = (const float*)d_in[4];
    sw      = (const float*)d_in[5];
    stab    = (const float*)d_in[6];
    tbW0    = (const float*)d_in[7];
    tbW1    = (const float*)d_in[8];
    tbb0    = (const float*)d_in[9];
    tbb1    = (const float*)d_in[10];
    Wv      = (const float*)d_in[11];
    mix0    = (const float*)d_in[12];
    for (int l = 0; l < 3; ++l) {
      int b = 13 + 8 * l;
      embW[l]  = (const float*)d_in[b + 0];
      embb[l]  = (const float*)d_in[b + 1];
      tpw[l]   = (const float*)d_in[b + 2];
      latW1[l] = (const float*)d_in[b + 3];
      latb1[l] = (const float*)d_in[b + 4];
      latW2[l] = (const float*)d_in[b + 5];
      latb2[l] = (const float*)d_in[b + 6];
      cmW[l]   = (const float*)d_in[b + 7];
    }
    N = in_sizes[0];
    E = in_sizes[1];
  } else {
    // Order B: jax pytree flatten (dict keys sorted recursively)
    dist    = (const float*)d_in[0];
    edst    = (const int*)d_in[1];
    esrc    = (const int*)d_in[2];
    Wv      = (const float*)d_in[3];
    for (int l = 0; l < 3; ++l) {
      int b = 4 + 8 * l;
      cmW[l]   = (const float*)d_in[b + 0];
      embW[l]  = (const float*)d_in[b + 1];
      embb[l]  = (const float*)d_in[b + 2];
      latW1[l] = (const float*)d_in[b + 3];
      latW2[l] = (const float*)d_in[b + 4];
      latb1[l] = (const float*)d_in[b + 5];
      latb2[l] = (const float*)d_in[b + 6];
      tpw[l]   = (const float*)d_in[b + 7];
    }
    mix0    = (const float*)d_in[28];
    stab    = (const float*)d_in[29];
    tbW0    = (const float*)d_in[30];
    tbW1    = (const float*)d_in[31];
    tbb0    = (const float*)d_in[32];
    tbb1    = (const float*)d_in[33];
    species = (const int*)d_in[34];
    sw      = (const float*)d_in[35];
    vecp    = (const float*)d_in[36];
    E = in_sizes[0];
    N = in_sizes[34];
  }

  float* xij = (float*)d_out;                 // E x 128
  float* Vij = xij + (size_t)E * 128;         // E x 16 x 9

  // workspace carve (floats)
  float* wsf = (float*)d_ws;
  size_t off = 0;
  float* enc   = wsf + off; off += (size_t)N * 16;
  float* Yb    = wsf + off; off += (size_t)E * 9;
  float* bufA  = wsf + off; off += (size_t)E * 128;   // feat (ld 48) / latent hidden
  float* bufB  = wsf + off; off += (size_t)E * 144;   // h1 (ld 128) / Lij (ld 144)
  float* bufC  = wsf + off; off += (size_t)E * 16;    // xw / rho_pre
  float* bufD  = wsf + off; off += (size_t)E * 16;    // scals
  float* dens  = wsf + off; off += (size_t)N * 144;
  int*   meta  = (int*)(wsf + off); off += 768;
  float* cgw   = wsf + off; off += (size_t)3 * 729 * 16;

  const int TPB = 256;
  int nbe  = (E + 127) / 128;                 // gemm blocks (128 edges each)

  cg_setup_kernel<<<1, TPB, 0, stream>>>(tpw[0], tpw[1], tpw[2], meta, cgw);
  enc_kernel<<<(N * 16 + TPB - 1) / TPB, TPB, 0, stream>>>(stab, species, enc, N);
  edge_geom_kernel<<<(E + TPB - 1) / TPB, TPB, 0, stream>>>(vecp, dist, enc, esrc, edst,
                                                            bufA, Yb, E);

  // two-body MLP: feat(40) -> 128 (silu) -> 128, * sw  => xij
  gemm_bf16_wmma<8, F_SILU><<<nbe, TPB, 0, stream>>>(
      bufA, 48, nullptr, 0, 40, 40, 64, tbW0, tbb0, nullptr, bufB, 128, E);
  gemm_bf16_wmma<8, F_MULSW><<<nbe, TPB, 0, stream>>>(
      bufB, 128, nullptr, 0, 128, 128, 128, tbW1, tbb1, sw, xij, 128, E);

  // xw = xij @ Wv ; Vij init
  gemm_bf16_wmma<1, 0><<<nbe, TPB, 0, stream>>>(
      xij, 128, nullptr, 0, 128, 128, 128, Wv, nullptr, nullptr, bufC, 16, E);
  vij_init_kernel<<<(E * 16 + TPB - 1) / TPB, TPB, 0, stream>>>(Yb, mix0, bufC, Vij, E);

  for (int l = 0; l < 3; ++l) {
    // rho_pre = (xij @ embW + embb) * sw
    gemm_bf16_wmma<1, F_MULSW><<<nbe, TPB, 0, stream>>>(
        xij, 128, nullptr, 0, 128, 128, 128, embW[l], embb[l], sw, bufC, 16, E);
    long dn = (long)N * 144;
    zero_kernel<<<(int)((dn + TPB - 1) / TPB), TPB, 0, stream>>>(dens, dn);
    scatter_kernel<<<(E * 16 + TPB - 1) / TPB, TPB, 0, stream>>>(bufC, Yb, esrc, dens, E);
    // Lij via sparse CG, scals = Lij[...,0]
    lij_kernel<<<(E + 15) / 16, TPB, 0, stream>>>(Vij, dens, esrc, meta,
                                                  cgw + (size_t)l * 729 * 16, bufB, bufD, E);
    // latent MLP: concat(xij, scals)(144) -> 128 (silu) -> 128; xij += out * sw
    gemm_bf16_wmma<8, F_SILU><<<nbe, TPB, 0, stream>>>(
        xij, 128, bufD, 16, 128, 144, 160, latW1[l], latb1[l], nullptr, bufA, 128, E);
    gemm_bf16_wmma<8, F_MULSW | F_ACC><<<nbe, TPB, 0, stream>>>(
        bufA, 128, nullptr, 0, 128, 128, 128, latW2[l], latb2[l], sw, xij, 128, E);
    // Vij = einsum('eci,oc->eoi', Lij, cm_W)
    vij_update_kernel<<<(E * 9 + TPB - 1) / TPB, TPB, 0, stream>>>(bufB, cmW[l], Vij, E);
  }
}
